// PointerNetwork_28578712388354
// MI455X (gfx1250) — matrix operand order repaired and checked by reading, not compiled, not run on hardware
//
#include <hip/hip_runtime.h>
#include <hip/hip_bf16.h>
#include <math.h>

#define S_LEN 32768
#define D_DIM 256
#define H_DIM 256
#define Z_DIM 1024  // 4*H
#define K2_LDS 64   // k-pairs staged in LDS (64 x 1024 x 4B = 256 KB)

typedef __attribute__((ext_vector_type(16))) _Float16 v16h;
typedef __attribute__((ext_vector_type(8)))  float    v8f;
typedef __attribute__((ext_vector_type(2)))  _Float16 h2v;

__device__ __forceinline__ float sigm(float x) { return 1.0f / (1.0f + __expf(-x)); }

// ---- CDNA5 wave32 WMMA fragment loaders (16x16x32 f16) ----
// A (16x32, row-major tile, row stride ld): lane l -> row l&15,
// K = {hi..hi+7} U {16+hi..16+hi+7}, hi = 8*(l>=16). Two contiguous 16B runs.
__device__ __forceinline__ v16h load_frag_a(const _Float16* tile, int ld) {
  int lane = threadIdx.x & 31;
  const _Float16* r = tile + (lane & 15) * ld + ((lane >> 4) << 3);
  v16h a;
#pragma unroll
  for (int i = 0; i < 8; ++i) { a[i] = r[i]; a[8 + i] = r[16 + i]; }
  return a;
}
// B (32x16) from an N-major (transposed) weight tile, row stride ld:
// lane l -> column l&15, K = {hi..hi+7} U {16+hi..16+hi+7}, hi = 8*(l>=16).
__device__ __forceinline__ v16h load_frag_bT(const _Float16* tileT, int ld) {
  int lane = threadIdx.x & 31;
  const _Float16* r = tileT + (lane & 15) * ld + ((lane >> 4) << 3);
  v16h b;
#pragma unroll
  for (int i = 0; i < 8; ++i) { b[i] = r[i]; b[8 + i] = r[16 + i]; }
  return b;
}

// ---- pre-pass: f16 copies, N-major transposes of Wx/We, k-paired Wh pack ----
__global__ void convert_kernel(const float* __restrict__ x, const float* __restrict__ Wx,
                               const float* __restrict__ WeW, const float* __restrict__ Wh,
                               _Float16* __restrict__ x16, _Float16* __restrict__ wxt,
                               _Float16* __restrict__ wet, h2v* __restrict__ whpk) {
  const long NX = (long)S_LEN * D_DIM;        // 8388608
  const long NW = (long)D_DIM * Z_DIM;        // 262144
  const long NE = (long)H_DIM * H_DIM;        // 65536
  const long NH = (long)(H_DIM / 2) * Z_DIM;  // 131072 packed pairs
  const long total = NX + NW + NE + NH;
  for (long i = (long)blockIdx.x * blockDim.x + threadIdx.x; i < total;
       i += (long)gridDim.x * blockDim.x) {
    if (i < NX) {
      x16[i] = (_Float16)x[i];
    } else if (i < NX + NW) {
      long t = i - NX; long n = t / D_DIM, k = t % D_DIM;
      wxt[t] = (_Float16)Wx[k * Z_DIM + n];          // wxt[n][k] = Wx[k][n]
    } else if (i < NX + NW + NE) {
      long t = i - NX - NW; long n = t / H_DIM, k = t % H_DIM;
      wet[t] = (_Float16)WeW[k * H_DIM + n];         // wet[n][k] = We[k][n]
    } else {
      long t = i - NX - NW - NE;                     // j fastest for coalescing
      long k2 = t / Z_DIM, j = t % Z_DIM;
      h2v w;
      w.x = (_Float16)Wh[(2 * k2) * Z_DIM + j];
      w.y = (_Float16)Wh[(2 * k2 + 1) * Z_DIM + j];
      whpk[t] = w;                                   // whpk[k2][j] = (Wh[2k2][j], Wh[2k2+1][j])
    }
  }
}

// ---- GEMM1: xwx[s, 0:1024] = x[s,:] @ enc_Wx + enc_b  (WMMA, bias in acc) ----
__global__ __launch_bounds__(256) void xwx_gemm_kernel(const _Float16* __restrict__ x16,
                                                       const _Float16* __restrict__ wxt,
                                                       const float* __restrict__ enc_b,
                                                       _Float16* __restrict__ xwx) {
  int lane = threadIdx.x & 31;
  int wave = threadIdx.x >> 5;           // 8 waves, each owns 128 output cols
  int row0 = blockIdx.x * 16;            // 16-row stripe per block
  v16h a[8];
#pragma unroll
  for (int kt = 0; kt < 8; ++kt)
    a[kt] = load_frag_a(x16 + (long)row0 * D_DIM + kt * 32, D_DIM);
#pragma unroll
  for (int ct = 0; ct < 8; ++ct) {
    int n0 = wave * 128 + ct * 16;
    int col = n0 + (lane & 15);
    float bv = enc_b[col];
    v8f c = {bv, bv, bv, bv, bv, bv, bv, bv};
#pragma unroll
    for (int kt = 0; kt < 8; ++kt) {
      v16h b = load_frag_bT(wxt + (long)n0 * D_DIM + kt * 32, D_DIM);
      c = __builtin_amdgcn_wmma_f32_16x16x32_f16(false, a[kt], false, b,
                                                 (short)0, c, false, false);
    }
    int rbase = row0 + ((lane >> 4) << 3);
#pragma unroll
    for (int r = 0; r < 8; ++r)
      xwx[(long)(rbase + r) * Z_DIM + col] = (_Float16)c[r];
  }
}

// ---- sequential LSTM scan: one persistent workgroup ----
// Per step: thread j computes z_j = xwx[t][j] + sum_k Wh[k][j] * h[k].
// CDNA5 WGP has 320 KB LDS: pin HALF the packed recurrent weights (256 KB)
// on-chip for the whole scan; the other half streams from the 192 MB L2.
// LDS weight reads are lane-consecutive b32 -> conflict-free across 64 banks.
__global__ __launch_bounds__(1024) void lstm_seq_kernel(const _Float16* __restrict__ xwx,
                                                        const h2v* __restrict__ whpk,
                                                        _Float16* __restrict__ enc16,
                                                        float* __restrict__ hf,
                                                        float* __restrict__ cf) {
  __shared__ h2v   wlds[K2_LDS * Z_DIM];   // 256 KB: k-pairs 0..63, all 1024 cols
  __shared__ float2 h2_lds[H_DIM / 2];
  __shared__ float z_lds[Z_DIM];
  int j = threadIdx.x;                     // output column 0..1023
  // stage first half of packed Wh into LDS (once)
  for (int idx = j; idx < K2_LDS * Z_DIM; idx += 1024) wlds[idx] = whpk[idx];
  if (j < H_DIM / 2) h2_lds[j] = make_float2(0.0f, 0.0f);
  float c = 0.0f, h = 0.0f;
  __syncthreads();
  for (int t = 0; t < S_LEN; ++t) {
    if (t + 1 < S_LEN)
      __builtin_prefetch(&xwx[(long)(t + 1) * Z_DIM + j], 0, 1);  // global_prefetch_b8
    float z = (float)xwx[(long)t * Z_DIM + j];   // includes bias
#pragma unroll 16
    for (int k2 = 0; k2 < K2_LDS; ++k2) {        // LDS-resident half
      h2v w = wlds[k2 * Z_DIM + j];
      float2 hp = h2_lds[k2];
      z = fmaf((float)w.x, hp.x, z);
      z = fmaf((float)w.y, hp.y, z);
    }
#pragma unroll 16
    for (int k2 = K2_LDS; k2 < H_DIM / 2; ++k2) {  // L2-streamed half
      h2v w = whpk[(long)k2 * Z_DIM + j];          // coalesced b32, 2 weights
      float2 hp = h2_lds[k2];
      z = fmaf((float)w.x, hp.x, z);
      z = fmaf((float)w.y, hp.y, z);
    }
    z_lds[j] = z;
    __syncthreads();                             // all h reads + z writes done
    if (j < H_DIM) {
      float zi = z_lds[j], zf = z_lds[H_DIM + j];
      float zg = z_lds[2 * H_DIM + j], zo = z_lds[3 * H_DIM + j];
      c = sigm(zf) * c + sigm(zi) * tanhf(zg);
      h = sigm(zo) * tanhf(c);
      ((float*)h2_lds)[j] = h;
      enc16[(long)t * H_DIM + j] = (_Float16)h;
    }
    __syncthreads();
  }
  if (j < H_DIM) { hf[j] = h; cf[j] = c; }
}

// ---- decoder step + dd[j] = (d @ Wd_W)[j] + Wd_b[j] + We_b[j] ----
__global__ __launch_bounds__(1024) void decoder_kernel(const float* __restrict__ st,
                                                       const float* __restrict__ dWx,
                                                       const float* __restrict__ dWh,
                                                       const float* __restrict__ db,
                                                       const float* __restrict__ hf,
                                                       const float* __restrict__ cf,
                                                       const float* __restrict__ WdW,
                                                       const float* __restrict__ Wdb,
                                                       const float* __restrict__ Web,
                                                       float* __restrict__ dd) {
  __shared__ float z_lds[Z_DIM];
  __shared__ float d_lds[H_DIM];
  int j = threadIdx.x;
  float z = db[j];
  for (int k = 0; k < D_DIM; ++k) z = fmaf(st[k], dWx[k * Z_DIM + j], z);
  for (int k = 0; k < H_DIM; ++k) z = fmaf(hf[k], dWh[k * Z_DIM + j], z);
  z_lds[j] = z;
  __syncthreads();
  if (j < H_DIM) {
    float zi = z_lds[j], zf = z_lds[H_DIM + j];
    float zg = z_lds[2 * H_DIM + j], zo = z_lds[3 * H_DIM + j];
    float c2 = sigm(zf) * cf[j] + sigm(zi) * tanhf(zg);
    d_lds[j] = sigm(zo) * tanhf(c2);
  }
  __syncthreads();
  if (j < H_DIM) {
    float acc = Web[j] + Wdb[j];
    for (int k = 0; k < H_DIM; ++k) acc = fmaf(d_lds[k], WdW[k * H_DIM + j], acc);
    dd[j] = acc;
  }
}

// ---- attention scores: WMMA GEMM + fused tanh/v epilogue + LDS row-reduce ----
__global__ __launch_bounds__(256) void attn_kernel(const _Float16* __restrict__ enc16,
                                                   const _Float16* __restrict__ wet,
                                                   const float* __restrict__ dd,
                                                   const float* __restrict__ vW,
                                                   const float* __restrict__ vb,
                                                   float* __restrict__ scores) {
  __shared__ float sc[16];
  int lane = threadIdx.x & 31;
  int wave = threadIdx.x >> 5;           // 8 waves x 2 tiles = 256 cols
  int row0 = blockIdx.x * 16;
  if (threadIdx.x < 16) sc[threadIdx.x] = 0.0f;
  __syncthreads();
  v16h a[8];
#pragma unroll
  for (int kt = 0; kt < 8; ++kt)
    a[kt] = load_frag_a(enc16 + (long)row0 * H_DIM + kt * 32, H_DIM);
#pragma unroll
  for (int ct = 0; ct < 2; ++ct) {
    int n0 = wave * 32 + ct * 16;
    int col = n0 + (lane & 15);
    v8f c = {};
#pragma unroll
    for (int kt = 0; kt < 8; ++kt) {
      v16h b = load_frag_bT(wet + (long)n0 * H_DIM + kt * 32, H_DIM);
      c = __builtin_amdgcn_wmma_f32_16x16x32_f16(false, a[kt], false, b,
                                                 (short)0, c, false, false);
    }
    float ddv = dd[col], vv = vW[col];
    int rbase = (lane >> 4) << 3;
#pragma unroll
    for (int r = 0; r < 8; ++r)
      atomicAdd(&sc[rbase + r], tanhf(c[r] + ddv) * vv);   // ds_add_f32
  }
  __syncthreads();
  if (threadIdx.x < 16) scores[row0 + threadIdx.x] = sc[threadIdx.x] + vb[0];
}

// ---- softmax over S in one block ----
__global__ __launch_bounds__(1024) void softmax_kernel(const float* __restrict__ scores,
                                                       float* __restrict__ out) {
  __shared__ float red[1024];
  int tid = threadIdx.x;
  float m = -INFINITY;
  for (int i = tid; i < S_LEN; i += 1024) m = fmaxf(m, scores[i]);
  red[tid] = m; __syncthreads();
  for (int s = 512; s > 0; s >>= 1) {
    if (tid < s) red[tid] = fmaxf(red[tid], red[tid + s]);
    __syncthreads();
  }
  float M = red[0]; __syncthreads();
  float sum = 0.0f;
  for (int i = tid; i < S_LEN; i += 1024) sum += __expf(scores[i] - M);
  red[tid] = sum; __syncthreads();
  for (int s = 512; s > 0; s >>= 1) {
    if (tid < s) red[tid] += red[tid + s];
    __syncthreads();
  }
  float inv = 1.0f / red[0];
  for (int i = tid; i < S_LEN; i += 1024) out[i] = __expf(scores[i] - M) * inv;
}

extern "C" void kernel_launch(void* const* d_in, const int* in_sizes, int n_in,
                              void* d_out, int out_size, void* d_ws, size_t ws_size,
                              hipStream_t stream) {
  const float* x      = (const float*)d_in[0];
  const float* enc_Wx = (const float*)d_in[1];
  const float* enc_Wh = (const float*)d_in[2];
  const float* enc_b  = (const float*)d_in[3];
  const float* dec_Wx = (const float*)d_in[4];
  const float* dec_Wh = (const float*)d_in[5];
  const float* dec_b  = (const float*)d_in[6];
  const float* We_W   = (const float*)d_in[7];
  const float* We_b   = (const float*)d_in[8];
  const float* Wd_W   = (const float*)d_in[9];
  const float* Wd_b   = (const float*)d_in[10];
  const float* v_W    = (const float*)d_in[11];
  const float* v_b    = (const float*)d_in[12];
  const float* start  = (const float*)d_in[13];
  float* out = (float*)d_out;
  (void)in_sizes; (void)n_in; (void)out_size; (void)ws_size;

  char* p = (char*)d_ws;
  _Float16* x16   = (_Float16*)p; p += (size_t)S_LEN * D_DIM * 2;       // 16 MB
  _Float16* wxt   = (_Float16*)p; p += (size_t)D_DIM * Z_DIM * 2;       // 0.5 MB
  _Float16* wet   = (_Float16*)p; p += (size_t)H_DIM * H_DIM * 2;       // 128 KB
  h2v*      whpk  = (h2v*)p;      p += (size_t)(H_DIM / 2) * Z_DIM * 4; // 512 KB
  _Float16* xwx   = (_Float16*)p; p += (size_t)S_LEN * Z_DIM * 2;       // 64 MB
  _Float16* enc16 = (_Float16*)p; p += (size_t)S_LEN * H_DIM * 2;       // 16 MB
  float* hf     = (float*)p; p += 1024;
  float* cf     = (float*)p; p += 1024;
  float* ddv    = (float*)p; p += 1024;
  float* scores = (float*)p; p += (size_t)S_LEN * 4;                    // 128 KB

  convert_kernel<<<4096, 256, 0, stream>>>(x, enc_Wx, We_W, enc_Wh, x16, wxt, wet, whpk);
  xwx_gemm_kernel<<<S_LEN / 16, 256, 0, stream>>>(x16, wxt, enc_b, xwx);
  lstm_seq_kernel<<<1, 1024, 0, stream>>>(xwx, whpk, enc16, hf, cf);
  decoder_kernel<<<1, 1024, 0, stream>>>(start, dec_Wx, dec_Wh, dec_b, hf, cf,
                                         Wd_W, Wd_b, We_b, ddv);
  attn_kernel<<<S_LEN / 16, 256, 0, stream>>>(enc16, wet, ddv, v_W, v_b, scores);
  softmax_kernel<<<1, 1024, 0, stream>>>(scores, out);
}
